// TrigoAttention_79791902425668
// MI455X (gfx1250) — matrix operand order repaired and checked by reading, not compiled, not run on hardware
//
#include <hip/hip_runtime.h>

typedef _Float16 v8h  __attribute__((ext_vector_type(8)));
typedef _Float16 v16h __attribute__((ext_vector_type(16)));
typedef float    v8f  __attribute__((ext_vector_type(8)));

#define D_MODEL 512
#define SEQ     128
#define BS      2
#define NTOK    (BS * SEQ)   // 256
#define HEADS   8
#define DK      64

#define INV2PI  0.15915494309189535f   // 1/(2*pi): V_SIN_F32 computes sin(2*pi*x)

// ---------------------------------------------------------------------------
// Weight pre-pack: pk[o*512+i] = (w0, w1/2pi, b_inner/2pi, 0).
// One b128 load then feeds each sin; 1/2pi prescale folded so the inner loop
// can use raw V_SIN_F32 (3 VALU ops per sin: fma, v_sin, fmac).
// ---------------------------------------------------------------------------
__global__ __launch_bounds__(256)
void trigo_pack_kernel(const float* __restrict__ w,   // [512,512,2]
                       const float* __restrict__ b,   // [512,513]
                       float4* __restrict__ pk)       // [512*512]
{
    int idx = blockIdx.x * 256 + threadIdx.x;          // 512*512 = 262144
    int o = idx >> 9, i = idx & 511;
    float2 ww = reinterpret_cast<const float2*>(w)[idx];
    float  bb = b[(size_t)o * (D_MODEL + 1) + i];
    pk[idx] = make_float4(ww.x, ww.y * INV2PI, bb * INV2PI, 0.0f);
}

// ---------------------------------------------------------------------------
// Trigonometric linear layer (instruction-issue optimized).
// Block = 256 threads: o = bx*32 + tid/8 (32 outputs), nl = tid&7;
// each thread accumulates rows nl and nl+8 of a 16-row token tile, so each
// packed-weight b128 load is amortized over 2 sins and broadcast across the
// 8 n-lanes sharing an o. LDS x-tile stride 516 dwords: 16B aligned (b128
// merges under unroll-4) and conflict-free (4*row mod 64 distinct).
// ---------------------------------------------------------------------------
__global__ __launch_bounds__(256)
void trigo_proj_kernel(const float*  __restrict__ x,    // [256,512]
                       const float4* __restrict__ pk,   // packed weights
                       const float*  __restrict__ b,    // [512,513] (final bias)
                       float* __restrict__ out32,       // [256,512] or null
                       _Float16* __restrict__ out16)    // [256,512] or null
{
    __shared__ float xs[16][516];
    const int tid = threadIdx.x;
    const int nl  = tid & 7;          // token lane (rows nl, nl+8)
    const int ol  = tid >> 3;         // output column within 32-wide tile
    const int n0  = blockIdx.y * 16;
    const int o   = blockIdx.x * 32 + ol;

    // cooperative x-tile load (vectorized, coalesced)
    {
        const float4* x4 = reinterpret_cast<const float4*>(x + (size_t)n0 * D_MODEL);
        for (int idx = tid; idx < 16 * (D_MODEL / 4); idx += 256) {
            int r = idx >> 7, c4 = idx & 127;
            *reinterpret_cast<float4*>(&xs[r][c4 * 4]) = x4[r * (D_MODEL / 4) + c4];
        }
    }
    __syncthreads();

    const float4* __restrict__ wp = pk + (size_t)o * D_MODEL;
    float acc0 = 0.0f, acc1 = 0.0f;
    #pragma unroll 4
    for (int i = 0; i < D_MODEL; ++i) {
        float4 w = wp[i];   // (w0, w1/2pi, b/2pi, -)
        acc0 += w.x * __builtin_amdgcn_sinf(fmaf(xs[nl][i],     w.y, w.z));
        acc1 += w.x * __builtin_amdgcn_sinf(fmaf(xs[nl + 8][i], w.y, w.z));
    }
    const float bias = b[(size_t)o * (D_MODEL + 1) + D_MODEL];
    acc0 += bias;
    acc1 += bias;

    const size_t i0 = (size_t)(n0 + nl) * D_MODEL + o;
    const size_t i1 = (size_t)(n0 + nl + 8) * D_MODEL + o;
    if (out32) { out32[i0] = acc0; out32[i1] = acc1; }
    if (out16) { out16[i0] = (_Float16)acc0; out16[i1] = (_Float16)acc1; }
}

// ---------------------------------------------------------------------------
// WMMA fragment helpers (wave32, 16x16x32 f16 shapes per CDNA5 ISA layouts).
// A (16x32): lane m=l%16, hi=l/16; halves t=0..15 <- K = kbase+16*hi+t
//                                   (contiguous, matching B).
// B (32x16): lane col N=l%16; halves t=0..15 <- K = kbase+16*hi+t (contiguous).
// C (16x16 f32): reg r, lane l -> row r+8*(l/16), col l%16.
// ---------------------------------------------------------------------------
__device__ inline v16h ldA(const _Float16* __restrict__ p, int hi) {
    v8h a0 = *(const v8h*)(p + 16 * hi);
    v8h a1 = *(const v8h*)(p + 16 * hi + 8);
    v16h r;
    #pragma unroll
    for (int t = 0; t < 8; ++t) { r[t] = a0[t]; r[8 + t] = a1[t]; }
    return r;
}
__device__ inline v16h ldB(const _Float16* __restrict__ p) {
    v8h a0 = *(const v8h*)p;
    v8h a1 = *(const v8h*)(p + 8);
    v16h r;
    #pragma unroll
    for (int t = 0; t < 8; ++t) { r[t] = a0[t]; r[8 + t] = a1[t]; }
    return r;
}

// ---------------------------------------------------------------------------
// Attention for one (b,h) per block. 256 threads = 8 waves; wave w owns
// q-rows [16w,16w+16). S=QK^T via WMMA, f32 softmax with shfl_xor reductions
// (masks 1..8 stay inside each 16-lane C-layout group), P through LDS to
// A-layout, O=PV via WMMA, softmax denominator folded in at the end.
// ---------------------------------------------------------------------------
__global__ __launch_bounds__(256)
void trigo_attn_kernel(const _Float16* __restrict__ qf,   // [256,512]
                       const _Float16* __restrict__ kf,
                       const _Float16* __restrict__ vf,
                       float* __restrict__ concat)        // [256,512]
{
    __shared__ _Float16 Vt[DK][136];      // V^T: [d][k], padded stride
    __shared__ _Float16 P16[SEQ][136];    // probs, row-major, padded

    const int tid  = threadIdx.x;
    const int bh   = blockIdx.x;
    const int b    = bh >> 3, h = bh & 7;
    const int wave = tid >> 5;
    const int lane = tid & 31;
    const int lo   = lane & 15, hi = lane >> 4;

    for (int idx = tid; idx < SEQ * DK; idx += 256) {
        int k = idx >> 6, d = idx & 63;
        Vt[d][k] = vf[(size_t)(b * SEQ + k) * D_MODEL + h * DK + d];
    }
    __syncthreads();

    // ---- S = (Q K^T) * 1/sqrt(dk) ----
    const _Float16* qrow =
        qf + (size_t)(b * SEQ + wave * 16 + lo) * D_MODEL + h * DK;
    const v16h aq0 = ldA(qrow, hi);        // d = 0..31
    const v16h aq1 = ldA(qrow + 32, hi);   // d = 32..63

    v8f st[8];
    #pragma unroll
    for (int kt = 0; kt < 8; ++kt) {
        const _Float16* krow =
            kf + (size_t)(b * SEQ + kt * 16 + lo) * D_MODEL + h * DK;
        v16h b0 = ldB(krow + 16 * hi);
        v16h b1 = ldB(krow + 32 + 16 * hi);
        v8f c = {0.f, 0.f, 0.f, 0.f, 0.f, 0.f, 0.f, 0.f};
        c = __builtin_amdgcn_wmma_f32_16x16x32_f16(false, aq0, false, b0,
                                                   (short)0, c, false, false);
        c = __builtin_amdgcn_wmma_f32_16x16x32_f16(false, aq1, false, b1,
                                                   (short)0, c, false, false);
        st[kt] = c;
    }

    // ---- softmax over 128 cols (per q-row) ----
    float mx[8], sm[8];
    #pragma unroll
    for (int r = 0; r < 8; ++r) mx[r] = -1e30f;
    #pragma unroll
    for (int kt = 0; kt < 8; ++kt)
        #pragma unroll
        for (int r = 0; r < 8; ++r) {
            float s = st[kt][r] * 0.125f;   // 1/sqrt(64)
            st[kt][r] = s;
            mx[r] = fmaxf(mx[r], s);
        }
    #pragma unroll
    for (int r = 0; r < 8; ++r) {
        float m = mx[r];
        m = fmaxf(m, __shfl_xor(m, 1, 32));
        m = fmaxf(m, __shfl_xor(m, 2, 32));
        m = fmaxf(m, __shfl_xor(m, 4, 32));
        m = fmaxf(m, __shfl_xor(m, 8, 32));
        mx[r] = m;
        sm[r] = 0.f;
    }
    #pragma unroll
    for (int kt = 0; kt < 8; ++kt)
        #pragma unroll
        for (int r = 0; r < 8; ++r) {
            float e = __expf(st[kt][r] - mx[r]);
            st[kt][r] = e;
            sm[r] += e;
        }
    #pragma unroll
    for (int r = 0; r < 8; ++r) {
        float s = sm[r];
        s += __shfl_xor(s, 1, 32);
        s += __shfl_xor(s, 2, 32);
        s += __shfl_xor(s, 4, 32);
        s += __shfl_xor(s, 8, 32);
        sm[r] = s;
    }

    // C-layout -> row-major f16 P in LDS (rows owned exclusively per wave)
    #pragma unroll
    for (int kt = 0; kt < 8; ++kt)
        #pragma unroll
        for (int r = 0; r < 8; ++r)
            P16[wave * 16 + r + 8 * hi][kt * 16 + lo] = (_Float16)st[kt][r];
    __syncthreads();

    // ---- O = P V, divide by row sums, write concat layout ----
    const _Float16* prow = &P16[wave * 16 + lo][0];
    #pragma unroll
    for (int dt = 0; dt < 4; ++dt) {
        v8f c = {0.f, 0.f, 0.f, 0.f, 0.f, 0.f, 0.f, 0.f};
        #pragma unroll
        for (int c4 = 0; c4 < 4; ++c4) {
            v16h a  = ldA(prow + 32 * c4, hi);
            v16h bb = ldB(&Vt[dt * 16 + lo][32 * c4 + 16 * hi]);
            c = __builtin_amdgcn_wmma_f32_16x16x32_f16(false, a, false, bb,
                                                       (short)0, c, false, false);
        }
        #pragma unroll
        for (int r = 0; r < 8; ++r) {
            const int n = b * SEQ + wave * 16 + r + 8 * hi;
            concat[(size_t)n * D_MODEL + h * DK + dt * 16 + lo] = c[r] / sm[r];
        }
    }
}

// ---------------------------------------------------------------------------
// Host-side orchestration (graph-capture safe: only kernel launches on stream)
// ---------------------------------------------------------------------------
extern "C" void kernel_launch(void* const* d_in, const int* in_sizes, int n_in,
                              void* d_out, int out_size, void* d_ws, size_t ws_size,
                              hipStream_t stream) {
    const float* q  = (const float*)d_in[0];
    const float* k  = (const float*)d_in[1];
    const float* v  = (const float*)d_in[2];
    const float* wq = (const float*)d_in[3];
    const float* bq = (const float*)d_in[4];
    const float* wk = (const float*)d_in[5];
    const float* bk = (const float*)d_in[6];
    const float* wv = (const float*)d_in[7];
    const float* bv = (const float*)d_in[8];
    const float* wo = (const float*)d_in[9];
    const float* bo = (const float*)d_in[10];

    // workspace: pk (4MB, reused per layer) | qf16 | kf16 | vf16 | concat(f32)
    float4* pk = (float4*)d_ws;
    _Float16* qf = (_Float16*)(pk + (size_t)D_MODEL * D_MODEL);
    _Float16* kf = qf + (size_t)NTOK * D_MODEL;
    _Float16* vf = kf + (size_t)NTOK * D_MODEL;
    float* concat = (float*)(vf + (size_t)NTOK * D_MODEL);

    const dim3 gpack((D_MODEL * D_MODEL) / 256);      // 1024 blocks
    const dim3 gproj(D_MODEL / 32, NTOK / 16);        // (16, 16) = 256 blocks

    trigo_pack_kernel<<<gpack, 256, 0, stream>>>(wq, bq, pk);
    trigo_proj_kernel<<<gproj, 256, 0, stream>>>(q, pk, bq, nullptr, qf);

    trigo_pack_kernel<<<gpack, 256, 0, stream>>>(wk, bk, pk);
    trigo_proj_kernel<<<gproj, 256, 0, stream>>>(k, pk, bk, nullptr, kf);

    trigo_pack_kernel<<<gpack, 256, 0, stream>>>(wv, bv, pk);
    trigo_proj_kernel<<<gproj, 256, 0, stream>>>(v, pk, bv, nullptr, vf);

    trigo_attn_kernel<<<BS * HEADS, 256, 0, stream>>>(qf, kf, vf, concat);

    trigo_pack_kernel<<<gpack, 256, 0, stream>>>(wo, bo, pk);
    trigo_proj_kernel<<<gproj, 256, 0, stream>>>(concat, pk, bo,
                                                 (float*)d_out, nullptr);
}
